// GAT_61383672594599
// MI455X (gfx1250) — matrix operand order repaired and checked by reading, not compiled, not run on hardware
//
#include <hip/hip_runtime.h>
#include <stdint.h>

// ---------------- problem constants ----------------
#define N_NODES 4096
#define FDIM    512
#define DDIM    64
#define HEADS   8
#define CDIM    40
#define CPAD    48        // pad classes to 3 WMMA n-tiles
#define LRELU_A 0.2f
#define NEGV    (-1e9f)

typedef __attribute__((ext_vector_type(16))) _Float16 v16h;
typedef __attribute__((ext_vector_type(8)))  _Float16 v8h;
typedef __attribute__((ext_vector_type(8)))  float    v8f;

// D = A(16x32 f16) * B(32x16 f16) + C(16x16 f32)
__device__ __forceinline__ v8f wmma_f16(v16h a, v16h b, v8f c) {
    return __builtin_amdgcn_wmma_f32_16x16x32_f16(
        false, a, false, b, (short)0, c, false, false);
}

// A-fragment: per lane (M = lane&15, half = lane>>4), K runs:
//   elements 0..7  -> K = half*8 + e ; elements 8..15 -> K = 16 + half*8 + (e-8)
__device__ __forceinline__ v16h load_frag_a(const _Float16* base, int half) {
    v8h lo = *(const v8h*)(base + half * 8);
    v8h hi = *(const v8h*)(base + 16 + half * 8);
    return __builtin_shufflevector(lo, hi, 0,1,2,3,4,5,6,7,8,9,10,11,12,13,14,15);
}

// B-fragment: per lane (N = lane&15, half = lane>>4), K runs:
//   elements 0..7  -> K = half*8 + e ; elements 8..15 -> K = 16 + half*8 + (e-8)
__device__ __forceinline__ v16h load_frag_b(const _Float16* base, int half) {
    v8h lo = *(const v8h*)(base + half * 8);
    v8h hi = *(const v8h*)(base + 16 + half * 8);
    return __builtin_shufflevector(lo, hi, 0,1,2,3,4,5,6,7,8,9,10,11,12,13,14,15);
}

// Async global->LDS 16B copy (per-lane addresses), tracked by ASYNCcnt.
__device__ __forceinline__ void async_copy_b128(uint32_t lds_off, const void* gptr) {
    asm volatile("global_load_async_to_lds_b128 %0, %1, off"
                 :: "v"(lds_off), "v"((uint64_t)(uintptr_t)gptr)
                 : "memory");
}
#define WAIT_ASYNC(n) asm volatile("s_wait_asynccnt %0" :: "i"(n) : "memory")

// ---------------------------------------------------------------------------
// Kernel 0: pre-convert operands to f16 for async (byte-copy) staging.
//   xb[i][k]  = f16(x[i][k])
//   W1t[h][d][k] = f16(W1[h][k][d])   (K-contiguous B operand)
// ---------------------------------------------------------------------------
__global__ void prep1_kernel(const float* __restrict__ x, const float* __restrict__ W1,
                             _Float16* __restrict__ xb, _Float16* __restrict__ W1t) {
    int idx = blockIdx.x * 256 + threadIdx.x;
    if (idx < N_NODES * FDIM) xb[idx] = (_Float16)x[idx];
    if (idx < HEADS * DDIM * FDIM) {
        int head = idx / (DDIM * FDIM);
        int rem  = idx % (DDIM * FDIM);
        int d = rem / FDIM, k = rem % FDIM;
        W1t[idx] = (_Float16)W1[((size_t)head * FDIM + k) * DDIM + d];
    }
}

// ---------------------------------------------------------------------------
// Kernel 1: h = x @ W1 per head.  Block = 256 thr (8 waves), tile 64x64.
// K-chunks of 32 double-buffer-staged into LDS via async b128 copies
// (one 16B async per thread per operand per chunk).  Output K-transposed:
// hbt[head][d][node] (f16) so attention B-fragment loads are b128s.
// ---------------------------------------------------------------------------
__global__ __launch_bounds__(256) void gemm1_kernel(
        const _Float16* __restrict__ xb, const _Float16* __restrict__ W1t,
        _Float16* __restrict__ hbt) {
    __shared__ _Float16 As[2][64][32];   // [buf][row][k]
    __shared__ _Float16 Bs[2][64][32];   // [buf][d][k]
    const int head = blockIdx.y;
    const int row0 = blockIdx.x * 64;
    const int tid  = threadIdx.x;
    const int wv   = tid >> 5, lane = tid & 31;
    const int m    = lane & 15, half = lane >> 4;
    const int mt   = wv & 3;             // 4 row tiles of 16
    const int nb   = (wv >> 2) * 32;     // waves 0-3: cols 0..31, 4-7: 32..63

    const int cr = tid >> 2;             // copy row (A: node, B: d)
    const int ck = (tid & 3) * 8;        // copy k-offset (8 f16 = 16 B)
    const _Float16* aSrc = xb + (size_t)(row0 + cr) * FDIM + ck;
    const _Float16* bSrc = W1t + ((size_t)head * DDIM + cr) * FDIM + ck;
    const uint32_t aDst = (uint32_t)(uintptr_t)&As[0][cr][ck];
    const uint32_t bDst = (uint32_t)(uintptr_t)&Bs[0][cr][ck];
    const uint32_t bstep = (uint32_t)sizeof(As[0]);   // 4096 bytes per buffer

    async_copy_b128(aDst, aSrc);
    async_copy_b128(bDst, bSrc);

    v8f acc0 = {}, acc1 = {};
    for (int i = 0; i < FDIM / 32; ++i) {
        if (i + 1 < FDIM / 32) {         // prefetch next chunk into other buffer
            const int kk = (i + 1) * 32;
            const uint32_t bo = ((i + 1) & 1) * bstep;
            async_copy_b128(aDst + bo, aSrc + kk);
            async_copy_b128(bDst + bo, bSrc + kk);
            WAIT_ASYNC(2);               // chunk i complete (in-order)
        } else {
            WAIT_ASYNC(0);
        }
        __syncthreads();
        const int b = i & 1;
        v16h af = load_frag_a(&As[b][mt * 16 + m][0], half);
        v16h b0 = load_frag_b(&Bs[b][nb + m][0],      half);
        v16h b1 = load_frag_b(&Bs[b][nb + 16 + m][0], half);
        acc0 = wmma_f16(af, b0, acc0);
        acc1 = wmma_f16(af, b1, acc1);
        __syncthreads();                 // readers done before buffer reuse
    }
    _Float16* outp = hbt + (size_t)head * DDIM * N_NODES;
    #pragma unroll
    for (int r = 0; r < 8; ++r) {
        int row = row0 + mt * 16 + r + half * 8;
        outp[(size_t)(nb + m) * N_NODES + row]      = (_Float16)acc0[r];
        outp[(size_t)(nb + 16 + m) * N_NODES + row] = (_Float16)acc1[r];
    }
}

// ---------------------------------------------------------------------------
// Kernel 2: si/sj = h . a1_{dst,src}, stored head-interleaved [node][8]
// ---------------------------------------------------------------------------
__global__ void sisj_kernel(const _Float16* __restrict__ hbt,
                            const float* __restrict__ a1_src,
                            const float* __restrict__ a1_dst,
                            float* __restrict__ siT, float* __restrict__ sjT) {
    int t = blockIdx.x * blockDim.x + threadIdx.x;
    if (t >= HEADS * N_NODES) return;
    int head = t / N_NODES, i = t % N_NODES;
    const _Float16* hb = hbt + (size_t)head * DDIM * N_NODES + i;
    float ssi = 0.f, ssj = 0.f;
    for (int d = 0; d < DDIM; ++d) {
        float v = (float)hb[(size_t)d * N_NODES];   // coalesced across i
        ssi += v * a1_dst[head * DDIM + d];
        ssj += v * a1_src[head * DDIM + d];
    }
    siT[i * HEADS + head] = ssi;
    sjT[i * HEADS + head] = ssj;
}

// ---------------------------------------------------------------------------
// Kernel 3 (template over #heads): online softmax stats per row, edges only
// (exact: every row has a self-loop, so non-edges underflow to exactly 0).
// ---------------------------------------------------------------------------
template <int HH>
__global__ __launch_bounds__(256) void stats_kernel(
        const uint8_t* __restrict__ adj,
        const float* __restrict__ siT, const float* __restrict__ sjT,
        float* __restrict__ mrow, float* __restrict__ lrow) {
    const int wv = threadIdx.x >> 5, lane = threadIdx.x & 31;
    const int i = blockIdx.x * 8 + wv;
    float si[HH], mx[HH], ls[HH];
    #pragma unroll
    for (int h = 0; h < HH; ++h) { si[h] = siT[i * HH + h]; mx[h] = NEGV; ls[h] = 0.f; }
    const uint8_t* arow = adj + (size_t)i * N_NODES;
    for (int j0 = lane * 4; j0 < N_NODES; j0 += 128) {
        uint32_t a4 = *(const uint32_t*)(arow + j0);
        if (a4) {
            #pragma unroll
            for (int b = 0; b < 4; ++b) {
                if ((a4 >> (8 * b)) & 0xffu) {
                    int j = j0 + b;
                    #pragma unroll
                    for (int h = 0; h < HH; ++h) {
                        float e = si[h] + sjT[j * HH + h];
                        e = fmaxf(e, LRELU_A * e);
                        float mn = fmaxf(mx[h], e);
                        ls[h] = ls[h] * __expf(mx[h] - mn) + __expf(e - mn);
                        mx[h] = mn;
                    }
                }
            }
        }
    }
    #pragma unroll
    for (int off = 16; off >= 1; off >>= 1) {
        #pragma unroll
        for (int h = 0; h < HH; ++h) {
            float mo = __shfl_xor(mx[h], off);
            float lo = __shfl_xor(ls[h], off);
            float mn = fmaxf(mx[h], mo);
            ls[h] = ls[h] * __expf(mx[h] - mn) + lo * __expf(mo - mn);
            mx[h] = mn;
        }
    }
    if (lane == 0) {
        #pragma unroll
        for (int h = 0; h < HH; ++h) {
            mrow[i * HH + h] = mx[h];
            lrow[i * HH + h] = ls[h];
        }
    }
}

// ---------------------------------------------------------------------------
// Kernel 4: out[h] = softmax(P) @ h via WMMA.  The 64x32 h-chunk is async-
// double-buffer staged into LDS once per block (shared by 8 waves); the
// 16x32 P-fragment is synthesized in registers from adj + rank-1 scores.
// Empty adj blocks skip frag-build + WMMA (barriers stay uniform).
// Epilogue: /l, elu, write concat x2b[node][head*64+d] (f16).
// ---------------------------------------------------------------------------
__global__ __launch_bounds__(256) void attn1_kernel(
        const uint8_t* __restrict__ adj,
        const float* __restrict__ siT, const float* __restrict__ sjT,
        const float* __restrict__ mrow, const float* __restrict__ lrow,
        const _Float16* __restrict__ hbt, _Float16* __restrict__ x2b) {
    __shared__ _Float16 Hs[2][64][32];   // [buf][d][j]
    const int head = blockIdx.y;
    const int wv = threadIdx.x >> 5, lane = threadIdx.x & 31;
    const int i0 = blockIdx.x * 128 + wv * 16;
    const int n = lane & 15, half = lane >> 4;
    const int ri = i0 + n;                       // A-fragment row of this lane
    const float si_v = siT[ri * HEADS + head];
    const float m_v  = mrow[ri * HEADS + head];
    const uint8_t* arow = adj + (size_t)ri * N_NODES;

    const int cd = threadIdx.x >> 2;             // copy: d row
    const int cj = (threadIdx.x & 3) * 8;        // copy: j offset (16 B)
    const _Float16* hSrc = hbt + ((size_t)head * DDIM + cd) * N_NODES + cj;
    const uint32_t hDst = (uint32_t)(uintptr_t)&Hs[0][cd][cj];
    const uint32_t bstep = (uint32_t)sizeof(Hs[0]);

    v8f acc[4] = {};
    async_copy_b128(hDst, hSrc);
    for (int i = 0; i < N_NODES / 32; ++i) {
        const int j0 = i * 32;
        if (i + 1 < N_NODES / 32) {
            async_copy_b128(hDst + ((i + 1) & 1) * bstep, hSrc + j0 + 32);
            WAIT_ASYNC(1);
        } else {
            WAIT_ASYNC(0);
        }
        __syncthreads();
        uint64_t a0 = *(const uint64_t*)(arow + j0 + half * 8);
        uint64_t a1 = *(const uint64_t*)(arow + j0 + 16 + half * 8);
        if (__ballot((a0 | a1) != 0ull) != 0ull) {   // any edge in 16x32 block
            v16h af;
            #pragma unroll
            for (int e = 0; e < 8; ++e) {
                int k = half * 8 + e;
                float p = 0.f;
                if ((a0 >> (8 * e)) & 0xffull) {
                    float ev = si_v + sjT[(j0 + k) * HEADS + head];
                    ev = fmaxf(ev, LRELU_A * ev);
                    p = __expf(ev - m_v);
                }
                af[e] = (_Float16)p;
            }
            #pragma unroll
            for (int e = 0; e < 8; ++e) {
                int k = 16 + half * 8 + e;
                float p = 0.f;
                if ((a1 >> (8 * e)) & 0xffull) {
                    float ev = si_v + sjT[(j0 + k) * HEADS + head];
                    ev = fmaxf(ev, LRELU_A * ev);
                    p = __expf(ev - m_v);
                }
                af[8 + e] = (_Float16)p;
            }
            const int b = i & 1;
            #pragma unroll
            for (int nt = 0; nt < 4; ++nt) {
                v16h bf = load_frag_b(&Hs[b][nt * 16 + n][0], half);
                acc[nt] = wmma_f16(af, bf, acc[nt]);
            }
        }
        __syncthreads();
    }
    #pragma unroll
    for (int r = 0; r < 8; ++r) {
        int row = i0 + r + half * 8;
        float linv = 1.0f / lrow[row * HEADS + head];
        #pragma unroll
        for (int nt = 0; nt < 4; ++nt) {
            float v = acc[nt][r] * linv;
            v = v > 0.f ? v : (__expf(v) - 1.f);            // elu
            x2b[(size_t)row * (HEADS * DDIM) + head * DDIM + nt * 16 + n] = (_Float16)v;
        }
    }
}

// ---------------------------------------------------------------------------
// Kernel 5: W2 (512x40 f32) -> W2bt (48x512 f16, K-contiguous, zero-padded)
// ---------------------------------------------------------------------------
__global__ void prep_w2_kernel(const float* __restrict__ W2, _Float16* __restrict__ W2bt) {
    int idx = blockIdx.x * 256 + threadIdx.x;
    if (idx >= CPAD * FDIM) return;
    int c = idx / FDIM, k = idx % FDIM;
    float v = (c < CDIM) ? W2[k * CDIM + c] : 0.f;
    W2bt[idx] = (_Float16)v;
}

// ---------------------------------------------------------------------------
// Kernel 6: h2 = x2 @ W2.  One wave per 16-row tile, 3 n-tiles (48 cols).
// ---------------------------------------------------------------------------
__global__ __launch_bounds__(256) void gemm2_kernel(
        const _Float16* __restrict__ x2b, const _Float16* __restrict__ W2bt,
        float* __restrict__ h2, _Float16* __restrict__ h2bt) {
    const int wv = threadIdx.x >> 5, lane = threadIdx.x & 31;
    const int i0 = (blockIdx.x * 8 + wv) * 16;
    const int n = lane & 15, half = lane >> 4;
    v8f acc[3] = {};
    for (int kk = 0; kk < FDIM; kk += 32) {
        v16h af = load_frag_a(x2b + (size_t)(i0 + n) * FDIM + kk, half);
        #pragma unroll
        for (int nt = 0; nt < 3; ++nt) {
            v16h bf = load_frag_b(W2bt + (size_t)(nt * 16 + n) * FDIM + kk, half);
            acc[nt] = wmma_f16(af, bf, acc[nt]);
        }
    }
    #pragma unroll
    for (int r = 0; r < 8; ++r) {
        int row = i0 + r + half * 8;
        #pragma unroll
        for (int nt = 0; nt < 3; ++nt) {
            int c = nt * 16 + n;
            float v = acc[nt][r];
            if (c < CDIM) h2[(size_t)row * CDIM + c] = v;
            h2bt[(size_t)c * N_NODES + row] = (_Float16)v;   // padded cols are 0
        }
    }
}

// ---------------------------------------------------------------------------
// Kernel 7: layer-2 scores si2 = h2.a2_dst, sj2 = h2.a2_src
// ---------------------------------------------------------------------------
__global__ void dvec_kernel(const float* __restrict__ h2,
                            const float* __restrict__ a2_src,
                            const float* __restrict__ a2_dst,
                            float* __restrict__ d_src, float* __restrict__ d_dst) {
    int i = blockIdx.x * 256 + threadIdx.x;
    if (i >= N_NODES) return;
    float s = 0.f, d = 0.f;
    for (int c = 0; c < CDIM; ++c) {
        float v = h2[(size_t)i * CDIM + c];
        s += v * a2_src[c];
        d += v * a2_dst[c];
    }
    d_src[i] = s;
    d_dst[i] = d;
}

// ---------------------------------------------------------------------------
// Kernel 8: final out = softmax(P2) @ h2 (f32 result, [N][40]).
// h2bt is only 384 KB (L2-resident, shared by all blocks) -> direct b128
// global fragment loads, no LDS staging needed.
// ---------------------------------------------------------------------------
__global__ __launch_bounds__(256) void attn2_kernel(
        const uint8_t* __restrict__ adj,
        const float* __restrict__ si2, const float* __restrict__ sj2,
        const float* __restrict__ m2, const float* __restrict__ l2,
        const _Float16* __restrict__ h2bt, float* __restrict__ outp) {
    const int wv = threadIdx.x >> 5, lane = threadIdx.x & 31;
    const int i0 = blockIdx.x * 128 + wv * 16;
    const int n = lane & 15, half = lane >> 4;
    const int ri = i0 + n;
    const float si_v = si2[ri];
    const float m_v  = m2[ri];
    const uint8_t* arow = adj + (size_t)ri * N_NODES;
    v8f acc[3] = {};

    for (int j0 = 0; j0 < N_NODES; j0 += 32) {
        uint64_t a0 = *(const uint64_t*)(arow + j0 + half * 8);
        uint64_t a1 = *(const uint64_t*)(arow + j0 + 16 + half * 8);
        if (__ballot((a0 | a1) != 0ull) == 0ull) continue;
        v16h af;
        #pragma unroll
        for (int e = 0; e < 8; ++e) {
            int k = half * 8 + e;
            float p = 0.f;
            if ((a0 >> (8 * e)) & 0xffull) {
                float ev = si_v + sj2[j0 + k];
                ev = fmaxf(ev, LRELU_A * ev);
                p = __expf(ev - m_v);
            }
            af[e] = (_Float16)p;
        }
        #pragma unroll
        for (int e = 0; e < 8; ++e) {
            int k = 16 + half * 8 + e;
            float p = 0.f;
            if ((a1 >> (8 * e)) & 0xffull) {
                float ev = si_v + sj2[j0 + k];
                ev = fmaxf(ev, LRELU_A * ev);
                p = __expf(ev - m_v);
            }
            af[8 + e] = (_Float16)p;
        }
        #pragma unroll
        for (int nt = 0; nt < 3; ++nt) {
            v16h bf = load_frag_b(h2bt + (size_t)(nt * 16 + n) * N_NODES + j0, half);
            acc[nt] = wmma_f16(af, bf, acc[nt]);
        }
    }
    #pragma unroll
    for (int r = 0; r < 8; ++r) {
        int row = i0 + r + half * 8;
        float linv = 1.0f / l2[row];
        #pragma unroll
        for (int nt = 0; nt < 3; ++nt) {
            int c = nt * 16 + n;
            if (c < CDIM) outp[(size_t)row * CDIM + c] = acc[nt][r] * linv;
        }
    }
}

// ---------------------------------------------------------------------------
extern "C" void kernel_launch(void* const* d_in, const int* in_sizes, int n_in,
                              void* d_out, int out_size, void* d_ws, size_t ws_size,
                              hipStream_t stream) {
    (void)in_sizes; (void)n_in; (void)out_size; (void)ws_size;
    const float*   x      = (const float*)d_in[0];
    const uint8_t* adj    = (const uint8_t*)d_in[1];
    const float*   W1     = (const float*)d_in[2];
    const float*   a1_src = (const float*)d_in[3];
    const float*   a1_dst = (const float*)d_in[4];
    const float*   W2     = (const float*)d_in[5];
    const float*   a2_src = (const float*)d_in[6];
    const float*   a2_dst = (const float*)d_in[7];
    float* outp = (float*)d_out;

    char* p = (char*)d_ws;
    auto carve = [&](size_t bytes) {
        char* r = p;
        p += (bytes + 255) & ~(size_t)255;
        return r;
    };
    _Float16* xb   = (_Float16*)carve((size_t)N_NODES * FDIM * 2);          // 4 MB
    _Float16* W1t  = (_Float16*)carve((size_t)HEADS * DDIM * FDIM * 2);     // 512 KB
    _Float16* hbt  = (_Float16*)carve((size_t)HEADS * DDIM * N_NODES * 2);  // 4 MB
    float*    siT  = (float*)   carve((size_t)N_NODES * HEADS * 4);
    float*    sjT  = (float*)   carve((size_t)N_NODES * HEADS * 4);
    float*    mrow = (float*)   carve((size_t)N_NODES * HEADS * 4);
    float*    lrow = (float*)   carve((size_t)N_NODES * HEADS * 4);
    _Float16* x2b  = (_Float16*)carve((size_t)N_NODES * HEADS * DDIM * 2);  // 4 MB
    _Float16* W2bt = (_Float16*)carve((size_t)CPAD * FDIM * 2);
    float*    h2   = (float*)   carve((size_t)N_NODES * CDIM * 4);
    _Float16* h2bt = (_Float16*)carve((size_t)CPAD * N_NODES * 2);
    float*    dsrc = (float*)   carve((size_t)N_NODES * 4);
    float*    ddst = (float*)   carve((size_t)N_NODES * 4);
    float*    m2   = (float*)   carve((size_t)N_NODES * 4);
    float*    l2   = (float*)   carve((size_t)N_NODES * 4);

    prep1_kernel<<<(N_NODES * FDIM + 255) / 256, 256, 0, stream>>>(x, W1, xb, W1t);
    gemm1_kernel<<<dim3(N_NODES / 64, HEADS), 256, 0, stream>>>(xb, W1t, hbt);
    sisj_kernel<<<(HEADS * N_NODES + 255) / 256, 256, 0, stream>>>(hbt, a1_src, a1_dst, siT, sjT);
    stats_kernel<HEADS><<<N_NODES / 8, 256, 0, stream>>>(adj, siT, sjT, mrow, lrow);
    attn1_kernel<<<dim3(N_NODES / 128, HEADS), 256, 0, stream>>>(adj, siT, sjT, mrow, lrow, hbt, x2b);
    prep_w2_kernel<<<(CPAD * FDIM + 255) / 256, 256, 0, stream>>>(W2, W2bt);
    gemm2_kernel<<<N_NODES / 16 / 8, 256, 0, stream>>>(x2b, W2bt, h2, h2bt);
    dvec_kernel<<<(N_NODES + 255) / 256, 256, 0, stream>>>(h2, a2_src, a2_dst, dsrc, ddst);
    stats_kernel<1><<<N_NODES / 8, 256, 0, stream>>>(adj, ddst, dsrc, m2, l2);
    attn2_kernel<<<N_NODES / 128, 256, 0, stream>>>(adj, ddst, dsrc, m2, l2, h2bt, outp);
}
